// Decoder_44049184588153
// MI455X (gfx1250) — 8-bit matrix operand order repaired and checked by reading, compiled, not run on hardware
//
#include <hip/hip_runtime.h>
#include <hip/hip_bf16.h>
#include <math.h>

#define T_STEPS 512
#define BATCH   256
#define ENC_HD  128
#define DEC_HD  128
#define NTHREADS 256
#define NWAVES   8

typedef int   v16i __attribute__((ext_vector_type(16)));
typedef float v8f  __attribute__((ext_vector_type(8)));

#if defined(__has_builtin)
#if __has_builtin(__builtin_amdgcn_cvt_pk_fp8_f32)
#define ATHENA_HW_FP8 1
#else
#define ATHENA_HW_FP8 0
#endif
#if __has_builtin(__builtin_amdgcn_tanhf)
#define ATHENA_HW_TANH 1
#else
#define ATHENA_HW_TANH 0
#endif
#else
#define ATHENA_HW_FP8 0
#define ATHENA_HW_TANH 0
#endif

static constexpr size_t OFF_WA1   = 0;
static constexpr size_t OFF_WHH   = OFF_WA1  + 128 * 384;
static constexpr size_t OFF_H     = OFF_WHH  + 512 * 128;
static constexpr size_t OFF_C     = OFF_H    + 256 * 128;
static constexpr size_t OFF_ENC   = OFF_C    + 256 * 128;
static constexpr size_t OFF_ENCBF = OFF_ENC  + 256 * 128;
static constexpr size_t OFF_F32   = OFF_ENCBF + 256 * 128 * 2;

struct SmallF32 {
    float score[256];
    float attw[256];
    float ctxPart[256];
    float ctx[128];
    float yt[256];
    float ba1[128];
    float wa2[128];
    float wfc[132];
    float bias[512];
    float wih[512];
    float wff[256];
    float red[64];
    float scal[8];
};
static constexpr size_t SMEM_BYTES = OFF_F32 + sizeof(SmallF32);

__device__ __forceinline__ unsigned char f32_to_fp8_sw(float x) {
    union { float f; unsigned u; } v; v.f = x;
    unsigned s   = (v.u >> 24) & 0x80u;
    unsigned abs = v.u & 0x7FFFFFFFu;
    if (abs > 0x43E00000u) return (unsigned char)(s | 0x7Eu);
    if (abs < 0x3C800000u) return (unsigned char)s;
    unsigned r = abs + 0x0007FFFFu + ((abs >> 20) & 1u);
    unsigned e = (r >> 23) - 120u;
    unsigned m = (r >> 20) & 7u;
    if (e > 15u) return (unsigned char)(s | 0x7Eu);
    return (unsigned char)(s | (e << 3) | m);
}
__device__ __forceinline__ unsigned char f32_to_fp8(float x) {
#if ATHENA_HW_FP8
    return (unsigned char)(__builtin_amdgcn_cvt_pk_fp8_f32(x, x, 0, false) & 0xFF);
#else
    return f32_to_fp8_sw(x);
#endif
}
__device__ __forceinline__ unsigned short pack2_fp8(float a, float b) {
#if ATHENA_HW_FP8
    return (unsigned short)(__builtin_amdgcn_cvt_pk_fp8_f32(a, b, 0, false) & 0xFFFF);
#else
    return (unsigned short)(f32_to_fp8_sw(a) | ((unsigned)f32_to_fp8_sw(b) << 8));
#endif
}
__device__ __forceinline__ unsigned short f32_to_bf16(float x) {
    union { float f; unsigned u; } v; v.f = x;
    unsigned r = v.u + 0x7FFFu + ((v.u >> 16) & 1u);
    return (unsigned short)(r >> 16);
}
__device__ __forceinline__ float bf16_to_f32(unsigned short h) {
    union { float f; unsigned u; } v; v.u = ((unsigned)h) << 16; return v.f;
}
__device__ __forceinline__ float fast_tanh(float x) {
#if ATHENA_HW_TANH
    return __builtin_amdgcn_tanhf(x);
#else
    return tanhf(x);
#endif
}
__device__ __forceinline__ float fast_sig(float x) {
    return 0.5f * fast_tanh(0.5f * x) + 0.5f;
}

__device__ __forceinline__ v16i load_a_fp8(const unsigned char* base, int rowStride,
                                           int m0, int lane) {
    int hl = lane >> 4;
    const unsigned char* rp = base + (m0 + (lane & 15)) * rowStride;
    v16i a;
#pragma unroll
    for (int blk = 0; blk < 2; ++blk) {
#pragma unroll
        for (int v = 0; v < 8; ++v) {
            int k = 64 * blk + 32 * (v >> 2) + 16 * hl + 8 * ((v >> 1) & 1) + 4 * (v & 1);
            a[8 * blk + v] = *(const int*)(rp + k);
        }
    }
    return a;
}
__device__ __forceinline__ v16i load_b_fp8(const unsigned char* base, int rowStride,
                                           int n0, int k0, int lane) {
    int hl = lane >> 4;
    const unsigned char* cp = base + (n0 + (lane & 15)) * rowStride + k0;
    v16i b;
#pragma unroll
    for (int v = 0; v < 16; ++v) {
        int k = 32 * (v >> 2) + 16 * hl + 4 * (v & 3);
        b[v] = *(const int*)(cp + k);
    }
    return b;
}
__device__ __forceinline__ v8f wmma_fp8(v16i a, v16i b, v8f c) {
    return __builtin_amdgcn_wmma_f32_16x16x128_fp8_fp8(a, b, (short)0, c, false, false);
}

__global__ __launch_bounds__(NTHREADS, 1)
void decoder_persistent(const float* __restrict__ enc,
                        const float* __restrict__ yhist,
                        const float* __restrict__ wa1, const float* __restrict__ ba1,
                        const float* __restrict__ wa2, const float* __restrict__ ba2,
                        const float* __restrict__ wfc, const float* __restrict__ bfc,
                        const float* __restrict__ wih, const float* __restrict__ bih,
                        const float* __restrict__ whh, const float* __restrict__ bhh,
                        const float* __restrict__ wff, const float* __restrict__ bff,
                        float* __restrict__ out)
{
    extern __shared__ unsigned char smem[];
    const int tid  = threadIdx.x;
    const int lane = tid & 31;
    const int wv   = tid >> 5;
    const int hl   = lane >> 4;

    unsigned char* sWa1 = smem + OFF_WA1;
    unsigned char* sWhh = smem + OFF_WHH;
    unsigned char* sH   = smem + OFF_H;
    unsigned char* sC   = smem + OFF_C;
    unsigned char* sEnc = smem + OFF_ENC;
    unsigned short* sEncBF = (unsigned short*)(smem + OFF_ENCBF);
    SmallF32& f = *(SmallF32*)(smem + OFF_F32);

    for (int i = tid; i < 128 * 384; i += NTHREADS) sWa1[i] = f32_to_fp8(wa1[i]);
    for (int i = tid; i < 512 * 128; i += NTHREADS) sWhh[i] = f32_to_fp8(whh[i]);
    for (int i = tid; i < 256 * 128; i += NTHREADS) { sH[i] = 0; sC[i] = 0; }
    if (tid < 128) { f.ba1[tid] = ba1[tid]; f.wa2[tid] = wa2[tid]; }
    if (tid < 132) f.wfc[tid] = (tid < 129) ? wfc[tid] : 0.0f;
    for (int i = tid; i < 512; i += NTHREADS) { f.bias[i] = bih[i] + bhh[i]; f.wih[i] = wih[i]; }
    f.wff[tid] = wff[tid];
    if (tid == 0) { f.scal[0] = ba2[0]; f.scal[1] = bfc[0]; f.scal[2] = bff[0]; }
    __syncthreads();

    float cReg[128];
#pragma unroll
    for (int i = 0; i < 128; ++i) cReg[i] = 0.0f;

    for (int t = 0; t < T_STEPS; ++t) {
        {
            const float2* ep2 = (const float2*)(enc + (size_t)t * BATCH * ENC_HD);
            for (int i = 0; i < 64; ++i) {
                int p = i * 256 + tid;
                float2 v = ep2[p];
                *(unsigned short*)(sEnc + 2 * p) = pack2_fp8(v.x, v.y);
                *(unsigned int*)((unsigned char*)sEncBF + 4 * p) =
                    (unsigned)f32_to_bf16(v.x) | ((unsigned)f32_to_bf16(v.y) << 16);
            }
            if (t + 1 < T_STEPS) {
                const float* np = enc + (size_t)(t + 1) * BATCH * ENC_HD + tid * 128;
#pragma unroll
                for (int i = 0; i < 4; ++i) __builtin_prefetch(np + i * 32, 0, 1);
            }
        }
        __syncthreads();

#pragma unroll
        for (int mtl = 0; mtl < 2; ++mtl) {
            int m0 = 32 * wv + 16 * mtl;
            v16i hA = load_a_fp8(sH,   128, m0, lane);
            v16i cA = load_a_fp8(sC,   128, m0, lane);
            v16i eA = load_a_fp8(sEnc, 128, m0, lane);
            float ps[8];
#pragma unroll
            for (int r = 0; r < 8; ++r) ps[r] = 0.0f;
            for (int nt = 0; nt < 8; ++nt) {
                v8f acc = {0.f,0.f,0.f,0.f,0.f,0.f,0.f,0.f};
                acc = wmma_fp8(hA, load_b_fp8(sWa1, 384, nt * 16, 0,   lane), acc);
                acc = wmma_fp8(cA, load_b_fp8(sWa1, 384, nt * 16, 128, lane), acc);
                acc = wmma_fp8(eA, load_b_fp8(sWa1, 384, nt * 16, 256, lane), acc);
                int n = nt * 16 + (lane & 15);
#pragma unroll
                for (int r = 0; r < 8; ++r)
                    ps[r] += f.wa2[n] * fast_tanh(acc[r] + f.ba1[n]);
            }
#pragma unroll
            for (int r = 0; r < 8; ++r) {
                ps[r] += __shfl_xor(ps[r], 1, 32);
                ps[r] += __shfl_xor(ps[r], 2, 32);
                ps[r] += __shfl_xor(ps[r], 4, 32);
                ps[r] += __shfl_xor(ps[r], 8, 32);
            }
            if ((lane & 15) == 0) {
#pragma unroll
                for (int r = 0; r < 8; ++r)
                    f.score[m0 + r + 8 * hl] = ps[r] + f.scal[0];
            }
        }
        __syncthreads();

        {
            float sv = f.score[tid];
            float mx = sv;
            for (int m = 1; m < 32; m <<= 1) mx = fmaxf(mx, __shfl_xor(mx, m, 32));
            if (lane == 0) f.red[wv] = mx;
            __syncthreads();
            mx = f.red[0];
#pragma unroll
            for (int i = 1; i < NWAVES; ++i) mx = fmaxf(mx, f.red[i]);
            float ex = __expf(sv - mx);
            float sm = ex;
            for (int m = 1; m < 32; m <<= 1) sm += __shfl_xor(sm, m, 32);
            if (lane == 0) f.red[8 + wv] = sm;
            __syncthreads();
            sm = 0.0f;
#pragma unroll
            for (int i = 0; i < NWAVES; ++i) sm += f.red[8 + i];
            f.attw[tid] = ex / sm;
        }
        __syncthreads();

        {
            int n = tid & 127, hb = (tid >> 7) * 128;
            float p = 0.0f;
            for (int b = 0; b < 128; ++b)
                p += f.attw[hb + b] * bf16_to_f32(sEncBF[(hb + b) * 128 + n]);
            f.ctxPart[tid] = p;
        }
        __syncthreads();
        if (tid < 128) f.ctx[tid] = f.ctxPart[tid] + f.ctxPart[tid + 128];
        __syncthreads();

        float Kff;
        {
            float Kt = f.scal[1];
            Kff = f.scal[2];
            for (int n = 0; n < 128; ++n) {
                float cv = f.ctx[n];
                Kt  += cv * f.wfc[n];
                Kff += cv * f.wff[128 + n];
            }
            float yv = yhist[(size_t)t * BATCH + tid];
            f.yt[tid] = Kt + f.wfc[128] * yv;
        }
        __syncthreads();

#pragma unroll
        for (int mtl = 0; mtl < 2; ++mtl) {
            int m0 = 32 * wv + 16 * mtl;
            v16i hA = load_a_fp8(sH, 128, m0, lane);
            float po[8];
#pragma unroll
            for (int r = 0; r < 8; ++r) po[r] = 0.0f;
            for (int nt = 0; nt < 8; ++nt) {
                v8f z = {0.f,0.f,0.f,0.f,0.f,0.f,0.f,0.f};
                v8f ai = wmma_fp8(hA, load_b_fp8(sWhh, 128,   0 + nt * 16, 0, lane), z);
                v8f af = wmma_fp8(hA, load_b_fp8(sWhh, 128, 128 + nt * 16, 0, lane), z);
                v8f ag = wmma_fp8(hA, load_b_fp8(sWhh, 128, 256 + nt * 16, 0, lane), z);
                v8f ao = wmma_fp8(hA, load_b_fp8(sWhh, 128, 384 + nt * 16, 0, lane), z);
                int jn = nt * 16 + (lane & 15);
#pragma unroll
                for (int r = 0; r < 8; ++r) {
                    int b = m0 + r + 8 * hl;
                    float ytb = f.yt[b];
                    float iv = fast_sig(ai[r] + f.bias[jn]       + ytb * f.wih[jn]);
                    float fv = fast_sig(af[r] + f.bias[128 + jn] + ytb * f.wih[128 + jn]);
                    float gv = fast_tanh(ag[r] + f.bias[256 + jn] + ytb * f.wih[256 + jn]);
                    float ov = fast_sig(ao[r] + f.bias[384 + jn] + ytb * f.wih[384 + jn]);
                    int ci = mtl * 64 + nt * 8 + r;
                    float cn = fv * cReg[ci] + iv * gv;
                    float hn = ov * fast_tanh(cn);
                    cReg[ci] = cn;
                    sH[b * 128 + jn] = f32_to_fp8(hn);
                    sC[b * 128 + jn] = f32_to_fp8(cn);
                    po[r] += f.wff[jn] * hn;
                }
            }
#pragma unroll
            for (int r = 0; r < 8; ++r) {
                po[r] += __shfl_xor(po[r], 1, 32);
                po[r] += __shfl_xor(po[r], 2, 32);
                po[r] += __shfl_xor(po[r], 4, 32);
                po[r] += __shfl_xor(po[r], 8, 32);
            }
            if ((lane & 15) == 0) {
#pragma unroll
                for (int r = 0; r < 8; ++r)
                    out[(size_t)t * BATCH + m0 + r + 8 * hl] = po[r] + Kff;
            }
        }
        __syncthreads();
    }
}

extern "C" void kernel_launch(void* const* d_in, const int* in_sizes, int n_in,
                              void* d_out, int out_size, void* d_ws, size_t ws_size,
                              hipStream_t stream) {
    (void)in_sizes; (void)n_in; (void)out_size; (void)d_ws; (void)ws_size;
    const float* enc   = (const float*)d_in[0];
    const float* yhist = (const float*)d_in[1];
    const float* wa1   = (const float*)d_in[2];
    const float* ba1   = (const float*)d_in[3];
    const float* wa2   = (const float*)d_in[4];
    const float* ba2   = (const float*)d_in[5];
    const float* wfc   = (const float*)d_in[6];
    const float* bfc   = (const float*)d_in[7];
    const float* wih   = (const float*)d_in[8];
    const float* bih   = (const float*)d_in[9];
    const float* whh   = (const float*)d_in[10];
    const float* bhh   = (const float*)d_in[11];
    const float* wff   = (const float*)d_in[12];
    const float* bff   = (const float*)d_in[13];
    float* out = (float*)d_out;

    decoder_persistent<<<dim3(1), dim3(NTHREADS), SMEM_BYTES, stream>>>(
        enc, yhist, wa1, ba1, wa2, ba2, wfc, bfc, wih, bih, whh, bhh, wff, bff, out);
}
